// BornCollapseSampler_13503377178930
// MI455X (gfx1250) — compile-verified, matrix operand order checked by reading, not run on hardware
//
#include <hip/hip_runtime.h>
#include <hip/hip_bf16.h>

// ---------------- problem constants ----------------
#define MROWS 256          // B*S = 32*8
#define KDIM  1024         // D
#define VDIM  50257        // vocab
#define LDAP  40           // padded A-row stride in bf16 (80B: 16B-aligned, conflict-free)
#define WPAD  36           // W LDS row stride in f32 (32 data + 4 pad dwords via TDM pad)

typedef __attribute__((ext_vector_type(16))) __bf16 v16bf;
typedef __attribute__((ext_vector_type(8)))  float  v8f;
typedef __attribute__((ext_vector_type(4)))  unsigned tdm_g0_t;
typedef __attribute__((ext_vector_type(8)))  int      tdm_g1_t;
typedef __attribute__((ext_vector_type(4)))  int      tdm_g2_t;
typedef __attribute__((ext_vector_type(8)))  int      tdm_g4_t;

union Frag { v16bf v; uint4 q[2]; unsigned u[8]; };

// ---- f32 pair -> packed bf16x2 in ONE v_perm_b32 (truncation rounding) ----
// result[15:0] = bf16(a), result[31:16] = bf16(b)
__device__ __forceinline__ unsigned pack2(float a, float b) {
  return __builtin_amdgcn_perm(__float_as_uint(b), __float_as_uint(a), 0x07060302u);
}

// ---- Tensor Data Mover: 2D f32 tile -> LDS, with 4-dword pad every 32 dwords ----
__device__ __forceinline__ void tdm_load_2d(const void* gptr, unsigned lds_off,
                                            unsigned tensor_d0, unsigned tensor_d1,
                                            unsigned tile_d0,   unsigned tile_d1,
                                            unsigned stride_d0)
{
#if defined(__has_builtin) && __has_builtin(__builtin_amdgcn_tensor_load_to_lds)
  unsigned long long ga = (unsigned long long)gptr;
  tdm_g0_t g0;
  g0[0] = 1u;                                   // count=1, user descriptor
  g0[1] = lds_off;                              // lds_addr
  g0[2] = (unsigned)ga;                         // global_addr[31:0]
  g0[3] = (unsigned)((ga >> 32) & 0x1FFFFFFu)   // global_addr[56:32]
        | (2u << 30);                           // type = 2 (image)
  tdm_g1_t g1;
  g1[0] = (int)((2u << 16)                      // data_size = 4B
              | (1u << 20)                      // pad_enable
              | (4u << 22)                      // pad_interval: 32 dwords
              | (3u << 25));                    // pad_amount:   4 dwords
  g1[1] = (int)((tensor_d0 & 0xFFFFu) << 16);                       // tensor_dim0 lo
  g1[2] = (int)((tensor_d0 >> 16) | ((tensor_d1 & 0xFFFFu) << 16)); // dim0 hi | dim1 lo
  g1[3] = (int)((tensor_d1 >> 16) | (tile_d0 << 16));               // dim1 hi | tile_dim0
  g1[4] = (int)tile_d1;                                             // tile_dim1 (tile_dim2=0)
  g1[5] = (int)stride_d0;                                           // tensor_dim0_stride lo
  g1[6] = 0;
  g1[7] = 0;
  tdm_g2_t gz = {0, 0, 0, 0};
  tdm_g4_t gz8 = {0, 0, 0, 0, 0, 0, 0, 0};
  __builtin_amdgcn_tensor_load_to_lds(g0, g1, gz, gz, gz8, 0);
#endif
}

// ============================================================
// GEMM: logits[m, v] = sum_k psiR[m,k]*WR[v,k] + psiI[m,k]*WI[v,k] + bias[v]
// 256 threads (8 waves). Workgroup: 64 cols x 256 rows. K-step 32 (bf16 WMMA,
// f32 accumulate). Wave w: colGroup = w>>1 (16 cols), M-half = w&1 (8 M-tiles,
// 64 acc VGPRs -> no VGPR-MSB pressure). W streamed by TDM (double-buffered).
// ============================================================
__global__ __launch_bounds__(256) void gemm_logits_kernel(
    const float* __restrict__ psiR, const float* __restrict__ psiI,
    const float* __restrict__ WR,   const float* __restrict__ WI,
    const float* __restrict__ bias, float* __restrict__ out)
{
  __shared__ __align__(16) unsigned short A_r[MROWS * LDAP];
  __shared__ __align__(16) unsigned short A_i[MROWS * LDAP];
  __shared__ __align__(16) float Wbuf[2][2][64 * WPAD];   // [buffer][real/imag][tile]

  const int tid   = threadIdx.x;
  const int wave  = tid >> 5;
  const int lane  = tid & 31;
  const int hl    = lane >> 4;        // half: 0 or 1
  const int r16   = lane & 15;
  const int cgrp  = wave >> 1;        // 0..3 : which 16-col group
  const int mhalf = wave & 1;         // 0..1 : which half of the 256 rows
  const int colBase64 = blockIdx.x * 64;
  const int colBase   = colBase64 + cgrp * 16;
  const unsigned remCols = (unsigned)(VDIM - colBase64); // OOB rows load zero via TDM

  v8f acc[8];
#pragma unroll
  for (int i = 0; i < 8; ++i)
    acc[i] = (v8f){0.f, 0.f, 0.f, 0.f, 0.f, 0.f, 0.f, 0.f};

  // ---- preamble: TDM-load first W tiles into buffer 0 ----
  if (wave == 0) {
    tdm_load_2d(WR + (size_t)colBase64 * KDIM, (unsigned)(unsigned long long)&Wbuf[0][0][0],
                KDIM, remCols, 32, 64, KDIM);
    tdm_load_2d(WI + (size_t)colBase64 * KDIM, (unsigned)(unsigned long long)&Wbuf[0][1][0],
                KDIM, remCols, 32, 64, KDIM);
  }

  for (int k0 = 0; k0 < KDIM; k0 += 32) {
    const int bb = (k0 >> 5) & 1;
    __syncthreads();                 // prior readers of A LDS and Wbuf[bb^1] are done

    // ---- TDM: prefetch next K-slab of W into the other buffer ----
    if (wave == 0 && (k0 + 32) < KDIM) {
      tdm_load_2d(WR + (size_t)colBase64 * KDIM + (k0 + 32),
                  (unsigned)(unsigned long long)&Wbuf[bb ^ 1][0][0],
                  KDIM, remCols, 32, 64, KDIM);
      tdm_load_2d(WI + (size_t)colBase64 * KDIM + (k0 + 32),
                  (unsigned)(unsigned long long)&Wbuf[bb ^ 1][1][0],
                  KDIM, remCols, 32, 64, KDIM);
    }

    // ---- stage psi tiles (256 rows x 32 k) as bf16; 2048 float4 per matrix ----
#pragma unroll
    for (int i = 0; i < 8; ++i) {
      int j   = tid + i * 256;          // 0..2047
      int row = j >> 3;
      int ch  = j & 7;
      float4 fr = ((const float4*)(psiR + (size_t)row * KDIM + k0))[ch];
      float4 fi = ((const float4*)(psiI + (size_t)row * KDIM + k0))[ch];
      unsigned* dr = (unsigned*)&A_r[row * LDAP + ch * 4];
      dr[0] = pack2(fr.x, fr.y); dr[1] = pack2(fr.z, fr.w);
      unsigned* di = (unsigned*)&A_i[row * LDAP + ch * 4];
      di[0] = pack2(fi.x, fi.y); di[1] = pack2(fi.z, fi.w);
    }

    // ---- gate on current W tiles (wave 0 waits, barrier releases the rest) ----
    if (wave == 0) {
      if ((k0 + 32) < KDIM) __builtin_amdgcn_s_wait_tensorcnt((short)2);
      else                  __builtin_amdgcn_s_wait_tensorcnt((short)0);
    }
    __syncthreads();

    // ---- B fragments: read f32 from TDM-landed LDS, convert to bf16 ----
    Frag br, bi;
#pragma unroll
    for (int m2 = 0; m2 < 2; ++m2) {
      const float* pw = &Wbuf[bb][m2][(cgrp * 16 + r16) * WPAD];
      float4 c0 = *(const float4*)(pw + hl * 8);
      float4 c1 = *(const float4*)(pw + hl * 8 + 4);
      float4 c2 = *(const float4*)(pw + 16 + hl * 8);
      float4 c3 = *(const float4*)(pw + 16 + hl * 8 + 4);
      Frag& f = m2 ? bi : br;
      f.u[0] = pack2(c0.x, c0.y); f.u[1] = pack2(c0.z, c0.w);
      f.u[2] = pack2(c1.x, c1.y); f.u[3] = pack2(c1.z, c1.w);
      f.u[4] = pack2(c2.x, c2.y); f.u[5] = pack2(c2.z, c2.w);
      f.u[6] = pack2(c3.x, c3.y); f.u[7] = pack2(c3.z, c3.w);
    }

    // ---- software-pipelined M-tile loop: 16 WMMAs per K-step per wave ----
    const int mbase = mhalf * 8;       // this wave's first M-tile
    Frag arc, aic;
    {
      const unsigned short* pr = &A_r[(mbase * 16 + r16) * LDAP];
      const unsigned short* pi = &A_i[(mbase * 16 + r16) * LDAP];
      arc.q[0] = *(const uint4*)(pr + hl * 8);
      arc.q[1] = *(const uint4*)(pr + 16 + hl * 8);
      aic.q[0] = *(const uint4*)(pi + hl * 8);
      aic.q[1] = *(const uint4*)(pi + 16 + hl * 8);
    }
#pragma unroll
    for (int mt = 0; mt < 8; ++mt) {
      Frag arn, ain;
      if (mt < 7) {
        const unsigned short* pr = &A_r[((mbase + mt + 1) * 16 + r16) * LDAP];
        const unsigned short* pi = &A_i[((mbase + mt + 1) * 16 + r16) * LDAP];
        arn.q[0] = *(const uint4*)(pr + hl * 8);
        arn.q[1] = *(const uint4*)(pr + 16 + hl * 8);
        ain.q[0] = *(const uint4*)(pi + hl * 8);
        ain.q[1] = *(const uint4*)(pi + 16 + hl * 8);
      }
      acc[mt] = __builtin_amdgcn_wmma_f32_16x16x32_bf16(
          false, arc.v, false, br.v, (short)0, acc[mt], false, false);
      acc[mt] = __builtin_amdgcn_wmma_f32_16x16x32_bf16(
          false, aic.v, false, bi.v, (short)0, acc[mt], false, false);
      if (mt < 7) { arc = arn; aic = ain; }
    }
  }

  // ---- epilogue: lane r16 = N, vgpr g -> M = 16*(mbase+mt) + 8*hl + g ----
  const int col = colBase + r16;
  const bool ok = col < VDIM;
  const float bv = ok ? bias[col] : 0.f;
#pragma unroll
  for (int mt = 0; mt < 8; ++mt) {
#pragma unroll
    for (int g = 0; g < 8; ++g) {
      int m = (mhalf * 8 + mt) * 16 + hl * 8 + g;
      if (ok) out[(size_t)m * VDIM + col] = acc[mt][g] + bv;
    }
  }
}

// ============================================================
// Per-row head: log_softmax, top-k(50) radix-select, top-p(0.95),
// renormalized probs, deterministic categorical draw.
// ============================================================
__device__ __forceinline__ unsigned fkey(float f) {   // order-preserving key
  unsigned u = __float_as_uint(f);
  return u ^ ((u >> 31) ? 0xFFFFFFFFu : 0x80000000u);
}
__device__ __forceinline__ float fkey_inv(unsigned k) {
  unsigned u = (k & 0x80000000u) ? (k ^ 0x80000000u) : ~k;
  return __uint_as_float(u);
}

__global__ __launch_bounds__(256) void sampler_kernel(float* __restrict__ d_out)
{
  const int row = blockIdx.x;
  const int tid = threadIdx.x;
  const int NT  = 256;
  const size_t MV = (size_t)MROWS * VDIM;

  const float* lg = d_out + (size_t)row * VDIM;            // logits (from GEMM)
  float* tokens   = d_out + MV;                            // [MROWS]
  float* probs    = d_out + MV + MROWS + (size_t)row * VDIM;
  float* logp     = d_out + MV + MROWS + MV + (size_t)row * VDIM;

  __shared__ float    red[256];
  __shared__ unsigned hist[256];
  __shared__ float    svals[256];
  __shared__ int      sidx[256];
  __shared__ float    s2v[256];
  __shared__ int      s2i[256];
  __shared__ float    pk[256];
  __shared__ unsigned sc_prefix;
  __shared__ int      sc_want;
  __shared__ unsigned sc_cnt;
  __shared__ int      sc_nk;

  // ---------- phase 1: max, sum-exp, log_softmax ----------
  float mx = -3.4e38f;
  for (int v = tid; v < VDIM; v += NT) mx = fmaxf(mx, lg[v]);
  red[tid] = mx; __syncthreads();
  for (int s = 128; s > 0; s >>= 1) {
    if (tid < s) red[tid] = fmaxf(red[tid], red[tid + s]);
    __syncthreads();
  }
  const float rmax = red[0]; __syncthreads();

  float sm = 0.f;
  for (int v = tid; v < VDIM; v += NT) sm += __expf(lg[v] - rmax);
  red[tid] = sm; __syncthreads();
  for (int s = 128; s > 0; s >>= 1) {
    if (tid < s) red[tid] += red[tid + s];
    __syncthreads();
  }
  const float lse = rmax + __logf(red[0]); __syncthreads();
  for (int v = tid; v < VDIM; v += NT) logp[v] = lg[v] - lse;

  // ---------- phase 2: exact 50th-largest via 4-pass radix select ----------
  if (tid == 0) { sc_prefix = 0u; sc_want = 50; }
  for (int shift = 24; shift >= 0; shift -= 8) {
    hist[tid] = 0u;
    __syncthreads();
    const unsigned pfx   = sc_prefix;
    const unsigned pmask = (shift == 24) ? 0u : (0xFFFFFFFFu << (shift + 8));
    for (int v = tid; v < VDIM; v += NT) {
      unsigned k = fkey(lg[v]);
      if ((k & pmask) == (pfx & pmask)) atomicAdd(&hist[(k >> shift) & 255u], 1u);
    }
    __syncthreads();
    if (tid == 0) {
      int want = sc_want; unsigned accu = 0; int b = 0;
      for (int i = 255; i >= 0; --i) {
        unsigned c = hist[i];
        if (accu + c >= (unsigned)want) { b = i; sc_want = want - (int)accu; break; }
        accu += c;
      }
      sc_prefix = pfx | ((unsigned)b << shift);
    }
    __syncthreads();
  }
  const float thr = fkey_inv(sc_prefix);

  // ---------- phase 3: gather survivors (keep l >= thr) ----------
  if (tid == 0) sc_cnt = 0u;
  __syncthreads();
  for (int v = tid; v < VDIM; v += NT) {
    float x = lg[v];
    if (x >= thr) {
      unsigned i = atomicAdd(&sc_cnt, 1u);
      if (i < 256u) { svals[i] = x; sidx[i] = v; }
    }
  }
  __syncthreads();
  const int cnt = (sc_cnt < 256u) ? (int)sc_cnt : 256;

  // ---------- phase 4: stable descending rank-sort ----------
  if (tid < cnt) {
    float v0 = svals[tid]; int i0 = sidx[tid]; int r = 0;
    for (int j = 0; j < cnt; ++j) {
      float vj = svals[j]; int ij = sidx[j];
      if (vj > v0 || (vj == v0 && ij < i0)) ++r;
    }
    s2v[r] = v0; s2i[r] = i0;
  }
  __syncthreads();

  // ---------- phase 5: top-p prefix cut, renormalize, sample ----------
  if (tid == 0) {
    const float m0 = s2v[0];
    float S = 0.f;
    for (int i = 0; i < cnt; ++i) { pk[i] = __expf(s2v[i] - m0); S += pk[i]; }
    float cum = 0.f; int nk = cnt;
    for (int i = 0; i < cnt; ++i) {          // mask_i = (cum_{i-1} >= p), mask_0 = false
      if (i > 0 && cum >= 0.95f) { nk = i; break; }
      cum += pk[i] / S;
    }
    float Z = 0.f;
    for (int i = 0; i < nk; ++i) Z += pk[i];
    for (int i = 0; i < nk; ++i) pk[i] /= Z;
    sc_nk = nk;
    unsigned h = (unsigned)row * 0x9E3779B9u + 0x2545F491u;
    h ^= h >> 16; h *= 0x7FEB352Du; h ^= h >> 15; h *= 0x846CA68Bu; h ^= h >> 16;
    float u = (float)(h >> 8) * (1.0f / 16777216.0f);
    float c = 0.f; int tok = s2i[nk - 1];
    for (int i = 0; i < nk; ++i) { c += pk[i]; if (u <= c) { tok = s2i[i]; break; } }
    tokens[row] = (float)tok;
  }
  __syncthreads();

  // ---------- phase 6: scatter probs into zeroed row ----------
  for (int v = tid; v < VDIM; v += NT) probs[v] = 0.f;
  __syncthreads();
  if (tid < sc_nk) probs[s2i[tid]] = pk[tid];
}

// ============================================================
extern "C" void kernel_launch(void* const* d_in, const int* in_sizes, int n_in,
                              void* d_out, int out_size, void* d_ws, size_t ws_size,
                              hipStream_t stream) {
  (void)in_sizes; (void)n_in; (void)out_size; (void)d_ws; (void)ws_size;
  const float* psiR = (const float*)d_in[0];
  const float* psiI = (const float*)d_in[1];
  const float* WR   = (const float*)d_in[2];
  const float* WI   = (const float*)d_in[3];
  const float* bias = (const float*)d_in[4];
  float* out = (float*)d_out;

  const int nblk = (VDIM + 63) / 64;   // 786 column strips
  gemm_logits_kernel<<<nblk, 256, 0, stream>>>(psiR, psiI, WR, WI, bias, out);
  sampler_kernel<<<MROWS, 256, 0, stream>>>(out);
}
